// A2_Graph_MFN_A_68882685493473
// MI455X (gfx1250) — compile-verified, matrix operand order checked by reading, not run on hardware
//
#include <hip/hip_runtime.h>
#include <hip/hip_bf16.h>
#include <stdint.h>
#include <stddef.h>

#define BLOCK 256
#define NWAVES 8

typedef __attribute__((ext_vector_type(16))) __bf16 v16bf;
typedef __attribute__((ext_vector_type(8)))  float  v8f;

// ---- input index map (setup_inputs order: text,audio,video, then params leaves in
// sorted-key (jax pytree) order) ----
enum {
  IN_TEXT=0, IN_AUDIO=1, IN_VIDEO=2,
  ATT2_W1=3, ATT2_W2=4, ATT2_B1=5, ATT2_B2=6,
  EFF_W1=7, EFF_W2=8, EFF_W3=9, EFF_B1=10, EFF_B2=11, EFF_B3=12,
  G1_W1=13, G1_W2=14, G1_B1=15, G1_B2=16,
  G2_W1=17, G2_W2=18, G2_B1=19, G2_B2=20,
  LSTMA_WHH=21, LSTMA_WIH=22, LSTMA_BHH=23, LSTMA_BIH=24,
  LSTML_WHH=25, LSTML_WIH=26, LSTML_BHH=27, LSTML_BIH=28,
  LSTMV_WHH=29, LSTMV_WIH=30, LSTMV_BHH=31, LSTMV_BIH=32,
  NET01_W1=33, NET01_W2=34, NET01_B1=35, NET01_B2=36,
  NET012_W1=37, NET012_W2=38, NET012_B1=39, NET012_B2=40,
  NET02_W1=41, NET02_W2=42, NET02_B1=43, NET02_B2=44,
  NET12_W1=45, NET12_W2=46, NET12_B1=47, NET12_B2=48,
  OUT_W1=49, OUT_W2=50, OUT_B1=51, OUT_B2=52,
  TNET_W1=53, TNET_W2=54, TNET_B1=55, TNET_B2=56,
  TRA_W=57, TRA_B=58, TRL_W=59, TRL_B=60, TRV_W=61, TRV_B=62,
  NUM_IN=63
};

// ---- packed weight matrix table ----
enum {
  W_LSTML_WHH=0, W_LSTMA_WHH, W_LSTMV_WHH,
  W_TRL, W_TRA, W_TRV,
  W_EFF1, W_EFF2, W_EFF3,
  W_N01_1, W_N01_2, W_N02_1, W_N02_2, W_N12_1, W_N12_2,
  W_N012_1, W_N012_2, W_TNET_1, W_TNET_2,
  W_ATT2_1, W_ATT2_2, W_G1_1, W_G1_2, W_G2_1, W_G2_2,
  W_OUT1,
  W_LSTML_WIH, W_LSTMA_WIH, W_LSTMV_WIH,
  W_COUNT
};

struct Ptrs { const float* p[NUM_IN]; };
struct Offs { unsigned o[W_COUNT]; };
struct PackDesc { const float* src; int N, K, Npad, Kpad; unsigned dstOff; };
struct PackArgs { PackDesc d[W_COUNT]; };

// ---- helpers ----
__device__ __forceinline__ unsigned short f2bf(float f) {
  unsigned u = __float_as_uint(f);
  u = (u + 0x7FFFu + ((u >> 16) & 1u)) >> 16;   // RNE
  return (unsigned short)u;
}
__device__ __forceinline__ float bf2f(unsigned short h) {
  return __uint_as_float(((unsigned)h) << 16);
}
__device__ __forceinline__ float sigf(float x) { return 1.0f / (1.0f + __expf(-x)); }

union BFrag { uint4 q[2]; v16bf v; };
union Acc8  { v8f v; float f[8]; };

// A fragment: 16xKpad bf16 rows in LDS (row stride aStride ushorts).
// lane L: m=L&15, kh=L>>4; VGPR j holds bf16 pair at k = kt*32 + (j>>2)*16 + kh*8 + (j&3)*2
__device__ __forceinline__ v16bf load_a_frag(const unsigned short* aLds, int aStride, int kt) {
  int lane = threadIdx.x & 31;
  int m = lane & 15, kh = lane >> 4;
  const unsigned short* base = aLds + m * aStride + kt * 32 + kh * 8;
  BFrag fr;
  fr.q[0] = *reinterpret_cast<const uint4*>(base);
  fr.q[1] = *reinterpret_cast<const uint4*>(base + 16);
  return fr.v;
}
// B fragment: pre-packed 1KB block per (nt,kt) tile, 8 dwords per lane
__device__ __forceinline__ v16bf load_b_frag(const unsigned* bp) {
  int lane = threadIdx.x & 31;
  const uint4* p = reinterpret_cast<const uint4*>(bp + lane * 8);
  BFrag fr;
  fr.q[0] = p[0];
  fr.q[1] = p[1];
  return fr.v;
}

// generic workgroup GEMM: D(16 x Ntiles*16) = act(A(16xKpad) * Bpacked + bias), f32 accum.
// act: 0 none, 1 relu, 2 tanh, 3 sigmoid
__device__ void wg_gemm(const unsigned short* aLds, int aStride, int Kpad,
                        const unsigned* bPack, int Ntiles,
                        float* outF, int outStride,
                        const float* bias, const float* bias2, int Nreal, int act,
                        const float* initF, int initStride,
                        const unsigned short* initG, int initGStride,
                        unsigned short* outG, int outGStride)
{
  const int wave = threadIdx.x >> 5;
  const int lane = threadIdx.x & 31;
  const int Ktiles = Kpad >> 5;
  const int mBase = (lane >> 4) * 8;
  const int nCol = lane & 15;
  for (int nt = wave; nt < Ntiles; nt += NWAVES) {
    const int n = nt * 16 + nCol;
    Acc8 acc;
    if (initF) {
#pragma unroll
      for (int r = 0; r < 8; ++r) acc.f[r] = initF[(mBase + r) * initStride + n];
    } else if (initG) {
#pragma unroll
      for (int r = 0; r < 8; ++r) acc.f[r] = bf2f(initG[(size_t)(mBase + r) * initGStride + n]);
    } else {
#pragma unroll
      for (int r = 0; r < 8; ++r) acc.f[r] = 0.0f;
    }
    const unsigned* bp = bPack + (size_t)nt * Ktiles * 256;
    for (int kt = 0; kt < Ktiles; ++kt) {
      v16bf a = load_a_frag(aLds, aStride, kt);
      v16bf b = load_b_frag(bp + (size_t)kt * 256);
      acc.v = __builtin_amdgcn_wmma_f32_16x16x32_bf16(false, a, false, b,
                                                      (short)0, acc.v, false, false);
    }
    float bv = 0.0f;
    if (n < Nreal) {
      if (bias)  bv += bias[n];
      if (bias2) bv += bias2[n];
    }
#pragma unroll
    for (int r = 0; r < 8; ++r) {
      float v = (n < Nreal) ? (acc.f[r] + bv) : 0.0f;
      if (act == 1) v = fmaxf(v, 0.0f);
      else if (act == 2) v = tanhf(v);
      else if (act == 3) v = sigf(v);
      int m = mBase + r;
      if (outF) outF[m * outStride + n] = v;
      if (outG) outG[(size_t)m * outGStride + n] = f2bf(v);
    }
  }
}

// stage f32 (row stride srcStride) -> bf16 segment at [16][colOff..colOff+Kseg) of dst
// (row stride dstStride). k>=Kreal -> 0. optional per-row scale (scale[m*scaleStride]).
__device__ void stage_seg(unsigned short* dstBase, int dstStride, int colOff,
                          const float* src, int srcStride, int Kseg, int Kreal,
                          const float* scale, int scaleStride)
{
  for (int idx = threadIdx.x; idx < 16 * Kseg; idx += BLOCK) {
    int m = idx / Kseg, k = idx - m * Kseg;
    float v = (k < Kreal) ? src[m * srcStride + k] : 0.0f;
    if (scale) v *= scale[m * scaleStride];
    dstBase[m * dstStride + colOff + k] = f2bf(v);
  }
}

// ---- kernel 0: pack fp32 weights -> bf16 WMMA-B-fragment layout ----
__global__ __launch_bounds__(BLOCK) void pack_kernel(PackArgs pa, unsigned* dst, unsigned total) {
  unsigned u = blockIdx.x * BLOCK + threadIdx.x;
  if (u >= total) return;
  int mi = 0;
#pragma unroll 1
  for (int i = 0; i < W_COUNT; ++i) {
    unsigned cnt = (unsigned)(pa.d[i].Npad * pa.d[i].Kpad / 2);
    if (u >= pa.d[i].dstOff && u < pa.d[i].dstOff + cnt) { mi = i; break; }
  }
  PackDesc dd = pa.d[mi];
  unsigned rel = u - dd.dstOff;
  int Ktiles = dd.Kpad >> 5;
  unsigned blk = rel >> 8;
  int within = (int)(rel & 255u);
  int nt = (int)(blk / (unsigned)Ktiles);
  int kt = (int)(blk % (unsigned)Ktiles);
  int lane = within >> 3, j = within & 7;
  int n = nt * 16 + (lane & 15);
  int kh = lane >> 4;
  int k = kt * 32 + ((j >> 2) << 4) + kh * 8 + ((j & 3) << 1);
  unsigned short lo = (n < dd.N && k     < dd.K) ? f2bf(dd.src[(size_t)n * dd.K + k    ]) : (unsigned short)0;
  unsigned short hi = (n < dd.N && (k+1) < dd.K) ? f2bf(dd.src[(size_t)n * dd.K + k + 1]) : (unsigned short)0;
  dst[u] = (unsigned)lo | ((unsigned)hi << 16);
}

// ---- kernel 1: time-parallel zx = x @ Wih^T + bih + bhh (bf16 out) ----
__global__ __launch_bounds__(BLOCK) void zx_kernel(Ptrs P, const unsigned* packW, Offs O,
                                                   unsigned short* zxl, unsigned short* zxa,
                                                   unsigned short* zxv)
{
  __shared__ unsigned short axb[16 * 768];
  int bx = blockIdx.x;
  int mod = bx >> 11;          // 2048 blocks per modality (128 t * 16 btiles)
  int r = bx & 2047;
  int t = r >> 4;
  int b0 = (r & 15) * 16;
  const float* x; int D; const unsigned* bw; const float* bih; const float* bhh; unsigned short* zx;
  if (mod == 0)      { x = P.p[IN_TEXT];  D = 768; bw = packW + O.o[W_LSTML_WIH]; bih = P.p[LSTML_BIH]; bhh = P.p[LSTML_BHH]; zx = zxl; }
  else if (mod == 1) { x = P.p[IN_AUDIO]; D = 128; bw = packW + O.o[W_LSTMA_WIH]; bih = P.p[LSTMA_BIH]; bhh = P.p[LSTMA_BHH]; zx = zxa; }
  else               { x = P.p[IN_VIDEO]; D = 256; bw = packW + O.o[W_LSTMV_WIH]; bih = P.p[LSTMV_BIH]; bhh = P.p[LSTMV_BHH]; zx = zxv; }
  for (int idx = threadIdx.x; idx < 16 * D; idx += BLOCK) {
    int m = idx / D, k = idx - m * D;
    axb[m * D + k] = f2bf(x[((size_t)(b0 + m) * 128 + t) * D + k]);  // x is [B,T,D]
  }
  __syncthreads();
  wg_gemm(axb, D, D, bw, 32, nullptr, 0, bih, bhh, 512, 0,
          nullptr, 0, nullptr, 0,
          zx + ((size_t)t * 256 + b0) * 512, 512);
}

// ---- kernel 2: persistent recurrence, one WG per 16 batch rows ----
__global__ __launch_bounds__(BLOCK, 1) void rec_kernel(Ptrs P, Offs O, const unsigned* packW,
                                                       const unsigned short* zxl,
                                                       const unsigned short* zxa,
                                                       const unsigned short* zxv,
                                                       float* out)
{
  // state + scratch in LDS (~242 KB; CDNA5 allows 320 KB/WG)
  __shared__ float hS[3][16 * 128];
  __shared__ float cS[3][16 * 128];
  __shared__ float memS[16 * 128];
  __shared__ float sS[16 * 384];           // [sl|sa|sv] f32
  __shared__ float oS[5][16 * 128];        // o01,o02,o12,o012,att
  __shared__ float msmS[3][16 * 128];      // cHat, g1, g2
  __shared__ float effS[16 * 32];          // eff (19 real, padded to 32)
  __shared__ float fS[16 * 512];           // generic f32 GEMM output
  __shared__ unsigned short abS[16 * 896]; // generic bf16 A staging
  __shared__ unsigned short sbS[16 * 384]; // [sl|sa|sv] bf16
  __shared__ unsigned short trS[3][16 * 256]; // [h_prev|h_new] bf16 per modality

  const int b0 = blockIdx.x * 16;
  const unsigned* wWhh[3] = { packW + O.o[W_LSTML_WHH], packW + O.o[W_LSTMA_WHH], packW + O.o[W_LSTMV_WHH] };
  const unsigned short* zxm[3] = { zxl, zxa, zxv };
  const unsigned* wTr[3] = { packW + O.o[W_TRL], packW + O.o[W_TRA], packW + O.o[W_TRV] };
  const float* bTr[3] = { P.p[TRL_B], P.p[TRA_B], P.p[TRV_B] };
  const unsigned* wN1[3] = { packW + O.o[W_N01_1], packW + O.o[W_N02_1], packW + O.o[W_N12_1] };
  const unsigned* wN2[3] = { packW + O.o[W_N01_2], packW + O.o[W_N02_2], packW + O.o[W_N12_2] };
  const float* bN1[3] = { P.p[NET01_B1], P.p[NET02_B1], P.p[NET12_B1] };
  const float* bN2[3] = { P.p[NET01_B2], P.p[NET02_B2], P.p[NET12_B2] };
  const int pA[3] = {0, 0, 1}, pB[3] = {1, 2, 2};

  for (int i = threadIdx.x; i < 2048; i += BLOCK) {
    hS[0][i] = hS[1][i] = hS[2][i] = 0.0f;
    cS[0][i] = cS[1][i] = cS[2][i] = 0.0f;
    memS[i] = 0.0f;
  }
  __syncthreads();

#pragma clang loop unroll(disable)
  for (int t = 0; t < 128; ++t) {
    // ---- 3 LSTM cells: gates = zx[t] + h_prev @ Whh^T ----
#pragma clang loop unroll(disable)
    for (int mod = 0; mod < 3; ++mod) {
      stage_seg(trS[mod], 256, 0, hS[mod], 128, 128, 128, nullptr, 0); // h_prev bf16
      __syncthreads();
      wg_gemm(trS[mod], 256, 128, wWhh[mod], 32,
              fS, 512, nullptr, nullptr, 512, 0,
              nullptr, 0, zxm[mod] + ((size_t)t * 256 + b0) * 512, 512,
              nullptr, 0);
      __syncthreads();
      for (int idx = threadIdx.x; idx < 2048; idx += BLOCK) {
        int m = idx >> 7, k = idx & 127;
        float iv = fS[m * 512 + k], fv = fS[m * 512 + 128 + k];
        float gv = fS[m * 512 + 256 + k], ov = fS[m * 512 + 384 + k];
        float cn = sigf(fv) * cS[mod][idx] + sigf(iv) * tanhf(gv);
        float hn = sigf(ov) * tanhf(cn);
        cS[mod][idx] = cn; hS[mod][idx] = hn;
        trS[mod][m * 256 + 128 + k] = f2bf(hn);
      }
      __syncthreads();
    }

    // ---- transforms: s = relu(W [h_prev, h_new] + b) ----
    for (int mod = 0; mod < 3; ++mod)
      wg_gemm(trS[mod], 256, 256, wTr[mod], 8,
              sS + mod * 128, 384, bTr[mod], nullptr, 128, 1,
              nullptr, 0, nullptr, 0, sbS + mod * 128, 384);
    __syncthreads();

    // ---- eff = W3 W2 W1 [sl sa sv] (all linear) ----
    wg_gemm(sbS, 384, 384, packW + O.o[W_EFF1], 7, fS, 512, P.p[EFF_B1], nullptr, 100, 0,
            nullptr, 0, nullptr, 0, nullptr, 0);
    __syncthreads();
    stage_seg(abS, 128, 0, fS, 512, 128, 100, nullptr, 0);
    __syncthreads();
    wg_gemm(abS, 128, 128, packW + O.o[W_EFF2], 8, fS, 512, P.p[EFF_B2], nullptr, 128, 0,
            nullptr, 0, nullptr, 0, nullptr, 0);
    __syncthreads();
    stage_seg(abS, 128, 0, fS, 512, 128, 128, nullptr, 0);
    __syncthreads();
    wg_gemm(abS, 128, 128, packW + O.o[W_EFF3], 2, effS, 32, P.p[EFF_B3], nullptr, 19, 0,
            nullptr, 0, nullptr, 0, nullptr, 0);
    __syncthreads();

    // ---- pair nets o01/o02/o12 ----
#pragma clang loop unroll(disable)
    for (int p2 = 0; p2 < 3; ++p2) {
      stage_seg(abS, 256, 0,   sS + pA[p2] * 128, 384, 128, 128, effS + 2 * p2,     32);
      stage_seg(abS, 256, 128, sS + pB[p2] * 128, 384, 128, 128, effS + 2 * p2 + 1, 32);
      __syncthreads();
      wg_gemm(abS, 256, 256, wN1[p2], 7, fS, 512, bN1[p2], nullptr, 100, 0,
              nullptr, 0, nullptr, 0, nullptr, 0);
      __syncthreads();
      stage_seg(abS, 128, 0, fS, 512, 128, 100, nullptr, 0);
      __syncthreads();
      wg_gemm(abS, 128, 128, wN2[p2], 8, oS[p2], 128, bN2[p2], nullptr, 128, 0,
              nullptr, 0, nullptr, 0, nullptr, 0);
      __syncthreads();
    }

    // ---- net012 ----
    stage_seg(abS, 768, 0,   sS + 0,   384, 128, 128, effS + 6,  32);
    stage_seg(abS, 768, 128, sS + 128, 384, 128, 128, effS + 7,  32);
    stage_seg(abS, 768, 256, sS + 256, 384, 128, 128, effS + 8,  32);
    stage_seg(abS, 768, 384, oS[0], 128, 128, 128, effS + 9,  32);
    stage_seg(abS, 768, 512, oS[1], 128, 128, 128, effS + 10, 32);
    stage_seg(abS, 768, 640, oS[2], 128, 128, 128, effS + 11, 32);
    __syncthreads();
    wg_gemm(abS, 768, 768, packW + O.o[W_N012_1], 7, fS, 512, P.p[NET012_B1], nullptr, 100, 0,
            nullptr, 0, nullptr, 0, nullptr, 0);
    __syncthreads();
    stage_seg(abS, 128, 0, fS, 512, 128, 100, nullptr, 0);
    __syncthreads();
    wg_gemm(abS, 128, 128, packW + O.o[W_N012_2], 8, oS[3], 128, P.p[NET012_B2], nullptr, 128, 0,
            nullptr, 0, nullptr, 0, nullptr, 0);
    __syncthreads();

    // ---- tnet -> att ----
    stage_seg(abS, 896, 0,   sS + 0,   384, 128, 128, effS + 12, 32);
    stage_seg(abS, 896, 128, sS + 128, 384, 128, 128, effS + 13, 32);
    stage_seg(abS, 896, 256, sS + 256, 384, 128, 128, effS + 14, 32);
    stage_seg(abS, 896, 384, oS[0], 128, 128, 128, effS + 15, 32);
    stage_seg(abS, 896, 512, oS[1], 128, 128, 128, effS + 16, 32);
    stage_seg(abS, 896, 640, oS[2], 128, 128, 128, effS + 17, 32);
    stage_seg(abS, 896, 768, oS[3], 128, 128, 128, effS + 18, 32);
    __syncthreads();
    wg_gemm(abS, 896, 896, packW + O.o[W_TNET_1], 7, fS, 512, P.p[TNET_B1], nullptr, 100, 0,
            nullptr, 0, nullptr, 0, nullptr, 0);
    __syncthreads();
    stage_seg(abS, 128, 0, fS, 512, 128, 100, nullptr, 0);
    __syncthreads();
    wg_gemm(abS, 128, 128, packW + O.o[W_TNET_2], 8, oS[4], 128, P.p[TNET_B2], nullptr, 128, 0,
            nullptr, 0, nullptr, 0, nullptr, 0);
    __syncthreads();

    // ---- MSM: cHat = tanh(att2(att)) ----
    stage_seg(abS, 128, 0, oS[4], 128, 128, 128, nullptr, 0);
    __syncthreads();
    wg_gemm(abS, 128, 128, packW + O.o[W_ATT2_1], 16, fS, 512, P.p[ATT2_B1], nullptr, 256, 1,
            nullptr, 0, nullptr, 0, nullptr, 0);
    __syncthreads();
    stage_seg(abS, 256, 0, fS, 512, 256, 256, nullptr, 0);
    __syncthreads();
    wg_gemm(abS, 256, 256, packW + O.o[W_ATT2_2], 8, msmS[0], 128, P.p[ATT2_B2], nullptr, 128, 2,
            nullptr, 0, nullptr, 0, nullptr, 0);
    __syncthreads();
    // g1 = sigmoid(g1([att,mem]))
    stage_seg(abS, 256, 0,   oS[4], 128, 128, 128, nullptr, 0);
    stage_seg(abS, 256, 128, memS,  128, 128, 128, nullptr, 0);
    __syncthreads();
    wg_gemm(abS, 256, 256, packW + O.o[W_G1_1], 16, fS, 512, P.p[G1_B1], nullptr, 256, 1,
            nullptr, 0, nullptr, 0, nullptr, 0);
    __syncthreads();
    stage_seg(abS, 256, 0, fS, 512, 256, 256, nullptr, 0);
    __syncthreads();
    wg_gemm(abS, 256, 256, packW + O.o[W_G1_2], 8, msmS[1], 128, P.p[G1_B2], nullptr, 128, 3,
            nullptr, 0, nullptr, 0, nullptr, 0);
    __syncthreads();
    // g2
    stage_seg(abS, 256, 0,   oS[4], 128, 128, 128, nullptr, 0);
    stage_seg(abS, 256, 128, memS,  128, 128, 128, nullptr, 0);
    __syncthreads();
    wg_gemm(abS, 256, 256, packW + O.o[W_G2_1], 16, fS, 512, P.p[G2_B1], nullptr, 256, 1,
            nullptr, 0, nullptr, 0, nullptr, 0);
    __syncthreads();
    stage_seg(abS, 256, 0, fS, 512, 256, 256, nullptr, 0);
    __syncthreads();
    wg_gemm(abS, 256, 256, packW + O.o[W_G2_2], 8, msmS[2], 128, P.p[G2_B2], nullptr, 128, 3,
            nullptr, 0, nullptr, 0, nullptr, 0);
    __syncthreads();
    // mem = g1*mem + g2*cHat
    for (int idx = threadIdx.x; idx < 2048; idx += BLOCK)
      memS[idx] = msmS[1][idx] * memS[idx] + msmS[2][idx] * msmS[0][idx];
    __syncthreads();
  }

  // ---- output head: relu(W1 [h_l,h_a,h_v,mem] + b1) -> dot with W2 + b2 ----
  stage_seg(abS, 512, 0,   hS[0], 128, 128, 128, nullptr, 0);
  stage_seg(abS, 512, 128, hS[1], 128, 128, 128, nullptr, 0);
  stage_seg(abS, 512, 256, hS[2], 128, 128, 128, nullptr, 0);
  stage_seg(abS, 512, 384, memS,  128, 128, 128, nullptr, 0);
  __syncthreads();
  wg_gemm(abS, 512, 512, packW + O.o[W_OUT1], 16, fS, 512, P.p[OUT_B1], nullptr, 256, 1,
          nullptr, 0, nullptr, 0, nullptr, 0);
  __syncthreads();
  if (threadIdx.x < 16) {
    int m = threadIdx.x;
    float acc = P.p[OUT_B2][0];
    for (int k = 0; k < 256; ++k) acc += fS[m * 512 + k] * P.p[OUT_W2][k];
    out[b0 + m] = acc;
  }
}

// ---- host ----
extern "C" void kernel_launch(void* const* d_in, const int* in_sizes, int n_in,
                              void* d_out, int out_size, void* d_ws, size_t ws_size,
                              hipStream_t stream)
{
  (void)in_sizes; (void)out_size; (void)ws_size;
  Ptrs P{};
  int nn = n_in < NUM_IN ? n_in : NUM_IN;
  for (int i = 0; i < nn; ++i) P.p[i] = (const float*)d_in[i];

  static const int srcIdx[W_COUNT] = {
    LSTML_WHH, LSTMA_WHH, LSTMV_WHH,
    TRL_W, TRA_W, TRV_W,
    EFF_W1, EFF_W2, EFF_W3,
    NET01_W1, NET01_W2, NET02_W1, NET02_W2, NET12_W1, NET12_W2,
    NET012_W1, NET012_W2, TNET_W1, TNET_W2,
    ATT2_W1, ATT2_W2, G1_W1, G1_W2, G2_W1, G2_W2,
    OUT_W1,
    LSTML_WIH, LSTMA_WIH, LSTMV_WIH
  };
  static const int Nn[W_COUNT] = { 512,512,512, 128,128,128, 100,128,19,
                                   100,128, 100,128, 100,128, 100,128, 100,128,
                                   256,128, 256,128, 256,128, 256, 512,512,512 };
  static const int Kk[W_COUNT] = { 128,128,128, 256,256,256, 384,100,128,
                                   256,100, 256,100, 256,100, 768,100, 896,100,
                                   128,256, 256,256, 256,256, 512, 768,128,256 };
  PackArgs pa; Offs O;
  unsigned off = 0;
  for (int i = 0; i < W_COUNT; ++i) {
    int Npad = (Nn[i] + 15) & ~15;
    int Kpad = (Kk[i] + 31) & ~31;
    pa.d[i].src = P.p[srcIdx[i]];
    pa.d[i].N = Nn[i]; pa.d[i].K = Kk[i];
    pa.d[i].Npad = Npad; pa.d[i].Kpad = Kpad;
    pa.d[i].dstOff = off;
    O.o[i] = off;
    off += (unsigned)(Npad * Kpad / 2);
  }
  unsigned total = off;

  unsigned* packW = (unsigned*)d_ws;
  size_t zxBaseU = ((size_t)total + 255) & ~(size_t)255; // uint offset
  unsigned short* zxl = (unsigned short*)(packW + zxBaseU);
  const size_t zxElems = (size_t)128 * 256 * 512; // T*B*4DH bf16
  unsigned short* zxa = zxl + zxElems;
  unsigned short* zxv = zxa + zxElems;

  pack_kernel<<<(total + BLOCK - 1) / BLOCK, BLOCK, 0, stream>>>(pa, packW, total);
  zx_kernel<<<3 * 128 * 16, BLOCK, 0, stream>>>(P, packW, O, zxl, zxa, zxv);
  rec_kernel<<<16, BLOCK, 0, stream>>>(P, O, packW, zxl, zxa, zxv, (float*)d_out);
}
